// MANN_LSTMCell_80479097193248
// MI455X (gfx1250) — matrix operand order repaired and checked by reading, not compiled, not run on hardware
//
#include <hip/hip_runtime.h>
#include <hip/hip_bf16.h>

#define MEMN   32768
#define UNITS  128
#define BATCH  256
#define IN_DIM 512
#define ZDIM   512   // 4*UNITS
#define KLSTM  768   // IN_DIM + UNITS (r) + UNITS (h)

typedef __attribute__((ext_vector_type(2)))  float  v2f;
typedef __attribute__((ext_vector_type(8)))  float  v8f;
typedef __attribute__((ext_vector_type(16))) __bf16 v16bf;
typedef __attribute__((ext_vector_type(4)))  unsigned int u32x4;
typedef __attribute__((ext_vector_type(4)))  int i32x4;
typedef __attribute__((ext_vector_type(8)))  int i32x8;

#if defined(__gfx1250__) && __has_builtin(__builtin_amdgcn_tensor_load_to_lds) && \
    __has_builtin(__builtin_amdgcn_s_wait_tensorcnt)
#define USE_TDM 1
#else
#define USE_TDM 0
#endif

#if __has_include(<hip/amd_detail/amd_gfx1250_TDM.h>)
#define TDM_6ARG 1
#else
#define TDM_6ARG 0
#endif

__device__ __forceinline__ float sigmoidf_(float x) { return 1.0f / (1.0f + __expf(-x)); }

#if USE_TDM
// Issue one 2-D TDM tile load (rows x width f32, row stride in elements) global -> LDS.
// D# layout per CDNA5 ISA 8.3/8.4: group0 {count=1, lds_addr, global_addr, type=2},
// group1 {data_size=4B, tensor dims == tile dims (no OOB needed), dim0 stride}.
__device__ __forceinline__ void tdm_load_2d_f32(const float* gptr, void* ldsptr,
                                                unsigned width, unsigned rows,
                                                unsigned stride) {
  unsigned long long ga = (unsigned long long)gptr;
  unsigned lds = (unsigned)(unsigned long long)ldsptr;  // low 32 bits == LDS byte offset
  u32x4 g0;
  g0[0] = 1u;                                            // count=1, user descriptor
  g0[1] = lds;                                           // lds_addr
  g0[2] = (unsigned)(ga & 0xFFFFFFFFu);                  // global_addr[31:0]
  g0[3] = (unsigned)(((ga >> 32) & 0x01FFFFFFu) | (2u << 30));  // addr[56:32] | type=2
  i32x8 g1;
  g1[0] = 0x00020000;                  // workgroup_mask=0, data_size=2 (4 bytes)
  g1[1] = (int)(width << 16);          // tensor_dim0[15:0] @ bit48
  g1[2] = (int)(rows << 16);           // tensor_dim0[31:16]=0 | tensor_dim1[15:0] @ bit80
  g1[3] = (int)(width << 16);          // tensor_dim1[31:16]=0 | tile_dim0 @ bit112
  g1[4] = (int)rows;                   // tile_dim1 @ bit128, tile_dim2=0
  g1[5] = (int)stride;                 // tensor_dim0_stride[31:0] @ bit160
  g1[6] = 0;
  g1[7] = 0;
  i32x4 z4 = {};
#if TDM_6ARG
  i32x8 z8 = {};
  __builtin_amdgcn_tensor_load_to_lds(g0, g1, z4, z4, z8, 0);
#else
  __builtin_amdgcn_tensor_load_to_lds(g0, g1, z4, z4, 0);
#endif
}
#endif

// ---------------------------------------------------------------- K0: init
__global__ void k_init(float* o_read, unsigned long long* minpack) {
  int t = blockIdx.x * blockDim.x + threadIdx.x;
  if (t < BATCH * UNITS) o_read[t] = 0.0f;
  if (t < BATCH) minpack[t] = ~0ull;
}

// ---------------------------------------------------------------- K1: LSTM (f32 WMMA 16x16x4)
// Branch-free segment: A[m][*] from Abase (row stride astride), B[*][n] from Bbase (row stride ZDIM).
__device__ __forceinline__ void lstm_seg(const float* __restrict__ Abase, int astride,
                                         const float* __restrict__ Bbase, int klen,
                                         int m0, int ncol, int g, int l16, v8f acc[4]) {
  for (int k = 0; k < klen; k += 4) {
    v2f a;
    a[0] = Abase[(m0 + l16) * astride + k + g * 2 + 0];
    a[1] = Abase[(m0 + l16) * astride + k + g * 2 + 1];
#pragma unroll
    for (int q = 0; q < 4; ++q) {
      v2f b;
      b[0] = Bbase[(k + g * 2 + 0) * ZDIM + q * UNITS + ncol];
      b[1] = Bbase[(k + g * 2 + 1) * ZDIM + q * UNITS + ncol];
      acc[q] = __builtin_amdgcn_wmma_f32_16x16x4_f32(false, a, false, b, (short)0, acc[q], false, false);
    }
  }
}

__global__ __launch_bounds__(32) void k_lstm(
    const float* __restrict__ inputs, const float* __restrict__ r_tm1,
    const float* __restrict__ h_tm1, const float* __restrict__ c_tm1,
    const float* __restrict__ kernelW, const float* __restrict__ recW,
    const float* __restrict__ bias, float* __restrict__ o_h, float* __restrict__ o_c) {
  int m0 = blockIdx.x * 16;   // batch tile
  int u0 = blockIdx.y * 16;   // unit tile (within each gate)
  int lane = threadIdx.x;
  int g = lane >> 4, l16 = lane & 15;
  int ncol = u0 + l16;
  v8f acc[4] = {};
  // z = [inputs | r_tm1 | h_tm1] @ [kernel ; rec_kernel], split into 3 branch-free segments.
  lstm_seg(inputs, IN_DIM, kernelW,                      IN_DIM, m0, ncol, g, l16, acc);
  lstm_seg(r_tm1,  UNITS,  kernelW + IN_DIM * ZDIM,      UNITS,  m0, ncol, g, l16, acc);
  lstm_seg(h_tm1,  UNITS,  recW,                         UNITS,  m0, ncol, g, l16, acc);
#pragma unroll
  for (int j = 0; j < 8; ++j) {
    int m = m0 + g * 8 + j;
    int u = u0 + l16;
    float iv = sigmoidf_(acc[0][j] + bias[u]);
    float fv = sigmoidf_(acc[1][j] + bias[UNITS + u]);
    float gv = tanhf(acc[2][j] + bias[2 * UNITS + u]);
    float ov = sigmoidf_(acc[3][j] + bias[3 * UNITS + u]);
    float cn = fv * c_tm1[m * UNITS + u] + iv * gv;
    float hn = ov * tanhf(cn);
    o_c[m * UNITS + u] = cn;
    o_h[m * UNITS + u] = hn;
  }
}

// ---------------------------------------------------------------- K2: h panels (n_key^T bf16, h^T bf16)
__global__ __launch_bounds__(128) void k_hprep(const float* __restrict__ o_h,
                                               __bf16* __restrict__ nkeyT,
                                               __bf16* __restrict__ hT) {
  __shared__ float s[128];
  int b = blockIdx.x, u = threadIdx.x;
  float h = o_h[b * UNITS + u];
  s[u] = h * h;
  __syncthreads();
  for (int off = 64; off > 0; off >>= 1) {
    if (u < off) s[u] += s[u + off];
    __syncthreads();
  }
  float inv = rsqrtf(fmaxf(s[0], 1e-12f));
  nkeyT[b * UNITS + u] = (__bf16)(h * inv);   // (batch, units): n_key transposed, contiguous K
  hT[u * BATCH + b]    = (__bf16)h;           // (units, batch): h transposed, contiguous K
}

// ---------------------------------------------------------------- K3: m row inverse norms
__global__ __launch_bounds__(256) void k_mnorm(const float* __restrict__ m_tm1,
                                               float* __restrict__ invm) {
  int w = threadIdx.x >> 5, lane = threadIdx.x & 31;
  int m = blockIdx.x * 8 + w;
  float s = 0.0f;
#pragma unroll
  for (int e = 0; e < 4; ++e) {
    float x = m_tm1[m * UNITS + lane + 32 * e];
    s += x * x;
  }
#pragma unroll
  for (int off = 16; off > 0; off >>= 1) s += __shfl_xor(s, off);
  if (lane == 0) invm[m] = rsqrtf(fmaxf(s, 1e-12f));
}

// ---------------------------------------------------------------- K4: cos-sim GEMM + softmax + usage (fused)
__global__ __launch_bounds__(256) void k_cos_softmax_usage(
    const float* __restrict__ m_tm1, const float* __restrict__ invm,
    const __bf16* __restrict__ nkeyT,
    const float* __restrict__ c_wu_tm1, const float* __restrict__ c_wlu_tm1,
    const float* __restrict__ c_wr_tm1, const float* __restrict__ wgate,
    float* __restrict__ o_cwr, float* __restrict__ o_cwu, float* __restrict__ o_cww,
    unsigned long long* __restrict__ minpack) {
  __shared__ __bf16 Ap[16 * 128];
  __shared__ float cosb[16 * 256];
  __shared__ float red[16 * 16];
  __shared__ float rowmax[16];
  __shared__ float rowinv[16];

  int m0 = blockIdx.x * 16;
  int tid = threadIdx.x;

  // Prefetch the streamed usage tensors for this tile into L2 (global_prefetch_b8).
  {
    int r = tid >> 4;
    size_t off = (size_t)(m0 + r) * BATCH + (tid & 15) * 16;
    __builtin_prefetch(c_wu_tm1 + off, 0, 1);
    __builtin_prefetch(c_wr_tm1 + off, 0, 1);
    __builtin_prefetch(c_wlu_tm1 + off, 0, 1);
  }

  // Stage normalized memory rows as bf16 A-panel (16 x 128).
#pragma unroll
  for (int ii = 0; ii < 8; ++ii) {
    int i = tid * 8 + ii;            // 0..2047
    int r = i >> 7, cpos = i & 127;
    Ap[i] = (__bf16)(m_tm1[(size_t)(m0 + r) * UNITS + cpos] * invm[m0 + r]);
  }
  __syncthreads();

  // 8 waves: wave w covers batch columns [w*32, w*32+32): 2 accumulator tiles.
  int w = tid >> 5, lane = tid & 31, g = lane >> 4, l16 = lane & 15;
  int nbase = w * 32;
  v8f acc0 = {}, acc1 = {};
  for (int ks = 0; ks < 4; ++ks) {
    int kb = ks * 32;
    v16bf a;
#pragma unroll
    for (int j = 0; j < 8; ++j) {
      a[j]     = Ap[l16 * 128 + kb + g * 16 + j];
      a[8 + j] = Ap[l16 * 128 + kb + g * 16 + 8 + j];
    }
    const v16bf* b0p = (const v16bf*)(nkeyT + (size_t)(nbase + l16) * UNITS + kb + g * 16);
    const v16bf* b1p = (const v16bf*)(nkeyT + (size_t)(nbase + 16 + l16) * UNITS + kb + g * 16);
    acc0 = __builtin_amdgcn_wmma_f32_16x16x32_bf16(false, a, false, *b0p, (short)0, acc0, false, false);
    acc1 = __builtin_amdgcn_wmma_f32_16x16x32_bf16(false, a, false, *b1p, (short)0, acc1, false, false);
  }
#pragma unroll
  for (int j = 0; j < 8; ++j) {
    cosb[(g * 8 + j) * 256 + nbase + l16]      = acc0[j];
    cosb[(g * 8 + j) * 256 + nbase + 16 + l16] = acc1[j];
  }
  __syncthreads();

  // Row-wise (over 256 batch cols) stable softmax in LDS.
  {
    int r = tid >> 4, c = tid & 15;
    float mx = -1e30f;
    for (int q = 0; q < 16; ++q) mx = fmaxf(mx, cosb[r * 256 + q * 16 + c]);
    red[r * 16 + c] = mx;
  }
  __syncthreads();
  if ((tid & 15) == 0) {
    int r = tid >> 4;
    float mx = red[r * 16];
    for (int q = 1; q < 16; ++q) mx = fmaxf(mx, red[r * 16 + q]);
    rowmax[r] = mx;
  }
  __syncthreads();
  {
    int r = tid >> 4, c = tid & 15;
    float s = 0.0f;
    for (int q = 0; q < 16; ++q) {
      float e = __expf(cosb[r * 256 + q * 16 + c] - rowmax[r]);
      cosb[r * 256 + q * 16 + c] = e;
      s += e;
    }
    red[r * 16 + c] = s;
  }
  __syncthreads();
  if ((tid & 15) == 0) {
    int r = tid >> 4;
    float s = 0.0f;
    for (int q = 0; q < 16; ++q) s += red[r * 16 + q];
    rowinv[r] = 1.0f / s;
  }
  __syncthreads();
  // Coalesced scale + store c_wr (thread == batch column).
  {
    int b = tid;
    for (int r = 0; r < 16; ++r) {
      float v = cosb[r * 256 + b] * rowinv[r];
      cosb[r * 256 + b] = v;
      o_cwr[(size_t)(m0 + r) * BATCH + b] = v;
    }
  }
  // Fused c_ww / c_wu + per-column packed min/argmin (values > 0 -> bit order == float order;
  // low word = ~row so ties pick the largest index, matching top_k's last-of-equal).
  {
    int b = tid;
    float wgv = sigmoidf_(wgate[0]);
    unsigned bfb = 0xFFFFFFFFu;
    int bidx = -1;
    for (int r = 0; r < 16; ++r) {
      int row = m0 + r;
      size_t i = (size_t)row * BATCH + b;
      float cww = wgv * c_wr_tm1[i] + (1.0f - wgv) + c_wlu_tm1[i];
      o_cww[i] = cww;
      float cwu = 0.95f * c_wu_tm1[i] + cosb[r * 256 + b] + cww;
      o_cwu[i] = cwu;
      unsigned fb = __float_as_uint(cwu);
      if (fb < bfb || (fb == bfb && row > bidx)) { bfb = fb; bidx = row; }
    }
    unsigned long long pack =
        ((unsigned long long)bfb << 32) | (unsigned long long)(0xFFFFFFFFu - (unsigned)bidx);
    atomicMin(&minpack[b], pack);
  }
}

// ---------------------------------------------------------------- K5: unpack min/argmin
__global__ void k_unpack(const unsigned long long* __restrict__ minpack,
                         float* __restrict__ nthv, int* __restrict__ nthi) {
  int b = threadIdx.x;
  unsigned long long p = minpack[b];
  nthv[b] = __uint_as_float((unsigned)(p >> 32));
  nthi[b] = (int)(0xFFFFFFFFu - (unsigned)(p & 0xFFFFFFFFull));
}

// ---------------------------------------------------------------- K6: c_wlu + memory = c_ww@h + scale*m
__global__ __launch_bounds__(256) void k_wlu_memory(
    const float* __restrict__ m_tm1, const float* __restrict__ o_cwu,
    const float* __restrict__ o_cww, const float* __restrict__ nthv,
    const int* __restrict__ nthi, const __bf16* __restrict__ hT,
    float* __restrict__ o_cwlu, float* __restrict__ o_mem) {
  __shared__ float s_nthv[256];
  __shared__ int   s_nthi[256];
  __shared__ __bf16 Aww[16 * 256];
  __shared__ float s_scale[16];

  int m0 = blockIdx.x * 16;
  int tid = threadIdx.x;
  s_nthv[tid] = nthv[tid];
  s_nthi[tid] = nthi[tid];
  __syncthreads();

  // c_wlu (coalesced) + stage c_ww tile as bf16 A-panel.
  {
    int b = tid;
    for (int r = 0; r < 16; ++r) {
      size_t i = (size_t)(m0 + r) * BATCH + b;
      o_cwlu[i] = (o_cwu[i] <= s_nthv[b]) ? 1.0f : 0.0f;
      Aww[r * 256 + b] = (__bf16)o_cww[i];
    }
  }
  // erase scale: (zv @ ones)[m,:] == 256 - #{b : nth_i[b]==m}
  if (tid < 16) {
    int cnt = 0;
    for (int b2 = 0; b2 < 256; ++b2) cnt += (s_nthi[b2] == m0 + tid) ? 1 : 0;
    s_scale[tid] = 256.0f - (float)cnt;
  }
  __syncthreads();

  int w = tid >> 5, lane = tid & 31, g = lane >> 4, l16 = lane & 15;
  int u0 = w * 16;
  v8f acc = {};
  for (int ks = 0; ks < 8; ++ks) {
    int kb = ks * 32;
    v16bf a;
#pragma unroll
    for (int j = 0; j < 8; ++j) {
      a[j]     = Aww[l16 * 256 + kb + g * 16 + j];
      a[8 + j] = Aww[l16 * 256 + kb + g * 16 + 8 + j];
    }
    const v16bf* bp = (const v16bf*)(hT + (size_t)(u0 + l16) * BATCH + kb + g * 16);
    acc = __builtin_amdgcn_wmma_f32_16x16x32_bf16(false, a, false, *bp, (short)0, acc, false, false);
  }
#pragma unroll
  for (int j = 0; j < 8; ++j) {
    int ml = g * 8 + j;
    int row = m0 + ml, u = u0 + l16;
    o_mem[(size_t)row * UNITS + u] = acc[j] + s_scale[ml] * m_tm1[(size_t)row * UNITS + u];
  }
}

// ---------------------------------------------------------------- K7: read = c_wr^T @ m_tm1 (split-K)
__global__ __launch_bounds__(512) void k_read_gemm(
    const float* __restrict__ o_cwr, const float* __restrict__ m_tm1,
    float* __restrict__ o_read) {
  __shared__ __bf16 AT[256 * 32];  // c_wr^T chunk (bf16): [batch][k]
  __shared__ __bf16 BT[128 * 32];  // m^T chunk (bf16):    [unit][k]
#if USE_TDM
  __shared__ float fA[32 * 256];   // raw c_wr chunk [k][b], filled by TDM
  __shared__ float fB[32 * 128];   // raw m chunk    [k][u], filled by TDM
#endif
  int tid = threadIdx.x;
  int w = tid >> 5, lane = tid & 31, g = lane >> 4, l16 = lane & 15;
  int b0 = w * 16;                  // 16 waves -> 256 batch rows
  v8f acc[8] = {};
  int k0 = blockIdx.x * 512;        // split-K chunk of memory rows

  for (int kb = 0; kb < 512; kb += 32) {
#if USE_TDM
    // Tensor Data Mover: DMA both f32 tiles into LDS, tracked by TENSORcnt.
    if (w == 0) {
      tdm_load_2d_f32(o_cwr + (size_t)(k0 + kb) * BATCH, fA, BATCH, 32, BATCH);
      tdm_load_2d_f32(m_tm1 + (size_t)(k0 + kb) * UNITS, fB, UNITS, 32, UNITS);
      __builtin_amdgcn_s_wait_tensorcnt(0);
    }
    __syncthreads();
    // LDS f32 -> LDS bf16 transpose.
#pragma unroll
    for (int q = 0; q < 16; ++q) {
      int i = tid + q * 512;        // 0..8191
      int k = i >> 8, b = i & 255;
      AT[b * 32 + k] = (__bf16)fA[k * 256 + b];
    }
#pragma unroll
    for (int q = 0; q < 8; ++q) {
      int i = tid + q * 512;        // 0..4095
      int k = i >> 7, u = i & 127;
      BT[u * 32 + k] = (__bf16)fB[k * 128 + u];
    }
#else
    // Fallback: coalesced global reads, transposed bf16 stores into LDS.
#pragma unroll
    for (int q = 0; q < 16; ++q) {
      int i = tid + q * 512;
      int k = i >> 8, b = i & 255;
      AT[b * 32 + k] = (__bf16)o_cwr[(size_t)(k0 + kb + k) * BATCH + b];
    }
#pragma unroll
    for (int q = 0; q < 8; ++q) {
      int i = tid + q * 512;
      int k = i >> 7, u = i & 127;
      BT[u * 32 + k] = (__bf16)m_tm1[(size_t)(k0 + kb + k) * UNITS + u];
    }
#endif
    __syncthreads();
    v16bf a;
#pragma unroll
    for (int j = 0; j < 8; ++j) {
      a[j]     = AT[(b0 + l16) * 32 + g * 16 + j];
      a[8 + j] = AT[(b0 + l16) * 32 + g * 16 + 8 + j];
    }
#pragma unroll
    for (int ct = 0; ct < 8; ++ct) {
      const v16bf* bp = (const v16bf*)(BT + (ct * 16 + l16) * 32 + g * 16);
      acc[ct] = __builtin_amdgcn_wmma_f32_16x16x32_bf16(false, a, false, *bp, (short)0, acc[ct], false, false);
    }
    __syncthreads();
  }
#pragma unroll
  for (int ct = 0; ct < 8; ++ct)
#pragma unroll
    for (int j = 0; j < 8; ++j) {
      int b = b0 + g * 8 + j;
      int u = ct * 16 + l16;
      atomicAdd(&o_read[(size_t)b * UNITS + u], acc[ct][j]);
    }
}

// ---------------------------------------------------------------- launch
extern "C" void kernel_launch(void* const* d_in, const int* in_sizes, int n_in,
                              void* d_out, int out_size, void* d_ws, size_t ws_size,
                              hipStream_t stream) {
  (void)in_sizes; (void)n_in; (void)out_size; (void)ws_size;
  const float* inputs    = (const float*)d_in[0];
  const float* r_tm1     = (const float*)d_in[1];
  const float* m_tm1     = (const float*)d_in[2];
  const float* c_wu_tm1  = (const float*)d_in[3];
  const float* c_wlu_tm1 = (const float*)d_in[4];
  const float* c_wr_tm1  = (const float*)d_in[5];
  const float* c_ww_tm1  = (const float*)d_in[6]; (void)c_ww_tm1;  // unused by reference
  const float* h_tm1     = (const float*)d_in[7];
  const float* c_tm1     = (const float*)d_in[8];
  const float* kernelW   = (const float*)d_in[9];
  const float* recW      = (const float*)d_in[10];
  const float* bias      = (const float*)d_in[11];
  const float* wgate     = (const float*)d_in[12];

  float* out = (float*)d_out;
  float* o_read = out + 0;
  float* o_mem  = out + 32768;
  float* o_cwu  = out + 4227072;
  float* o_cwlu = out + 12615680;
  float* o_cwr  = out + 21004288;
  float* o_cww  = out + 29392896;
  float* o_h    = out + 37781504;
  float* o_c    = out + 37814272;

  char* ws = (char*)d_ws;
  unsigned long long* w_minpack = (unsigned long long*)(ws + 0);       // 256 * 8B
  float* w_nthv  = (float*)(ws + 2048);                                // 256 * 4B
  int*   w_nthi  = (int*)(ws + 3072);                                  // 256 * 4B
  float* w_invm  = (float*)(ws + 4096);                                // 32768 * 4B
  __bf16* w_nkeyT = (__bf16*)(ws + 135168);                            // 256*128 bf16
  __bf16* w_hT    = (__bf16*)(ws + 200704);                            // 128*256 bf16

  k_init<<<128, 256, 0, stream>>>(o_read, w_minpack);
  k_lstm<<<dim3(16, 8), 32, 0, stream>>>(inputs, r_tm1, h_tm1, c_tm1, kernelW, recW, bias, o_h, o_c);
  k_hprep<<<BATCH, 128, 0, stream>>>(o_h, w_nkeyT, w_hT);
  k_mnorm<<<MEMN / 8, 256, 0, stream>>>(m_tm1, w_invm);
  k_cos_softmax_usage<<<MEMN / 16, 256, 0, stream>>>(m_tm1, w_invm, w_nkeyT,
                                                     c_wu_tm1, c_wlu_tm1, c_wr_tm1, wgate,
                                                     o_cwr, o_cwu, o_cww, w_minpack);
  k_unpack<<<1, 256, 0, stream>>>(w_minpack, w_nthv, w_nthi);
  k_wlu_memory<<<MEMN / 16, 256, 0, stream>>>(m_tm1, o_cwu, o_cww, w_nthv, w_nthi, w_hT,
                                              o_cwlu, o_mem);
  k_read_gemm<<<MEMN / 512, 512, 0, stream>>>(o_cwr, m_tm1, o_read);
}
